// ScaledDotProductAttention_62062277427456
// MI455X (gfx1250) — matrix operand order repaired and checked by reading, not compiled, not run on hardware
//
#include <hip/hip_runtime.h>

typedef __attribute__((ext_vector_type(16))) __bf16 v16bf;
typedef __attribute__((ext_vector_type(8)))  __bf16 v8bf;
typedef __attribute__((ext_vector_type(8)))  float  v8f;
typedef __attribute__((ext_vector_type(4)))  float  v4f;

#define BATCH 32
#define SEQ   2048
#define DIM   64
#define QT    64            // queries per workgroup
#define KT    64            // keys per block iteration (4 x 16-key WMMA tiles)
#define NWAVE 4
#define MASKV (-4294967295.0f)   // -2^32 + 1
#define SCALE 0.125f             // 1/sqrt(64), folded into Q at load

__global__ __launch_bounds__(128, 2)
void fa_fwd_kernel(const float* __restrict__ Qg, const float* __restrict__ Kg,
                   const float* __restrict__ Vg, float* __restrict__ Og)
{
    __shared__ __align__(16) __bf16 Qs[QT][DIM];         // 8 KB, pre-scaled by 1/sqrt(d)
    __shared__ __align__(16) __bf16 Ks[KT][DIM];         // 8 KB (key, dim)
    __shared__ __align__(16) __bf16 Vt[DIM][KT];         // 8 KB (dim, key) transposed
    __shared__ __align__(16) float  Os[NWAVE][16][DIM];  // 16 KB f32 epilogue staging
    __shared__ __align__(16) float  kadd[KT];            // 0.0 or MASK_NUM (additive key mask)
    __shared__ float qmask[QT];

    const int tid  = threadIdx.x;
    const int wid  = tid >> 5;
    const int lane = tid & 31;
    const int h    = lane >> 4;   // half-wave selector
    const int ln   = lane & 15;

    const int b  = blockIdx.x / (SEQ / QT);
    const int q0 = (blockIdx.x % (SEQ / QT)) * QT;

    const float* Qb = Qg + (size_t)b * SEQ * DIM;
    const float* Kb = Kg + (size_t)b * SEQ * DIM;
    const float* Vb = Vg + (size_t)b * SEQ * DIM;

    // ---- Q tile: global f32 -> LDS bf16 (scaled), plus query-padding mask ----
    {
        const int row  = tid >> 1;
        const int half = (tid & 1) * 32;
        const float* src = Qb + (size_t)(q0 + row) * DIM + half;
        float asum = 0.f;
#pragma unroll
        for (int i = 0; i < 8; ++i) {
            v4f x = *(const v4f*)(src + i * 4);
            asum += __builtin_fabsf(x.x) + __builtin_fabsf(x.y) +
                    __builtin_fabsf(x.z) + __builtin_fabsf(x.w);
            Qs[row][half + i*4 + 0] = (__bf16)(x.x * SCALE);
            Qs[row][half + i*4 + 1] = (__bf16)(x.y * SCALE);
            Qs[row][half + i*4 + 2] = (__bf16)(x.z * SCALE);
            Qs[row][half + i*4 + 3] = (__bf16)(x.w * SCALE);
        }
        asum += __shfl_xor(asum, 1, 32);
        if ((tid & 1) == 0) qmask[row] = (asum != 0.f) ? 1.f : 0.f;
    }
    __syncthreads();

    // ---- loop-invariant B-fragments of Q^T (lane = query column, 16 contiguous dims) ----
    v16bf bq[2];
#pragma unroll
    for (int ks = 0; ks < 2; ++ks)
        bq[ks] = *(const v16bf*)&Qs[wid*16 + ln][ks*32 + h*16];

    // ---- per-lane (per-query) online softmax state; acc = O^T (row = dim, col = query) ----
    float m_run = -3.0e38f, l_run = 0.f;
    v8f acc[4];
#pragma unroll
    for (int t = 0; t < 4; ++t) { v8f z = {}; acc[t] = z; }

    const int qmin_w = q0 + wid * 16;        // smallest query this wave owns
    const int qmax_w = qmin_w + 15;
    const int vq     = qmin_w + ln;          // this lane's query index

    for (int kb = 0; kb < q0 + QT; kb += KT) {
        __syncthreads();   // previous iteration's LDS reads complete

        // ---- cooperative K/V block load: f32 -> bf16 LDS, additive key mask ----
        {
            const int row = tid >> 1;          // key in block (0..63)
            const int col = (tid & 1) * 32;    // 32 dims per thread
            const float* ksrc = Kb + (size_t)(kb + row) * DIM + col;
            const float* vsrc = Vb + (size_t)(kb + row) * DIM + col;
            float asum = 0.f;
#pragma unroll
            for (int i = 0; i < 8; ++i) {
                v4f x = *(const v4f*)(ksrc + i * 4);
                asum += __builtin_fabsf(x.x) + __builtin_fabsf(x.y) +
                        __builtin_fabsf(x.z) + __builtin_fabsf(x.w);
                Ks[row][col + i*4 + 0] = (__bf16)x.x;
                Ks[row][col + i*4 + 1] = (__bf16)x.y;
                Ks[row][col + i*4 + 2] = (__bf16)x.z;
                Ks[row][col + i*4 + 3] = (__bf16)x.w;
            }
            asum += __shfl_xor(asum, 1, 32);
            if ((tid & 1) == 0) kadd[row] = (asum != 0.f) ? 0.f : MASKV;
#pragma unroll
            for (int i = 0; i < 8; ++i) {
                v4f x = *(const v4f*)(vsrc + i * 4);
                Vt[col + i*4 + 0][row] = (__bf16)x.x;
                Vt[col + i*4 + 1][row] = (__bf16)x.y;
                Vt[col + i*4 + 2][row] = (__bf16)x.z;
                Vt[col + i*4 + 3][row] = (__bf16)x.w;
            }
            if (kb + KT < q0 + QT) {             // hide next block's latency
                __builtin_prefetch(ksrc + KT * DIM, 0, 1);
                __builtin_prefetch(vsrc + KT * DIM, 0, 1);
            }
        }
        __syncthreads();

        if (kb <= qmax_w) {
            // ---- S^T = K Q^T : 4 key tiles x 2 K-steps; A = K rows (contiguous) ----
            v8f ct[4];
#pragma unroll
            for (int kt = 0; kt < 4; ++kt) {
                v8f z = {};
                ct[kt] = z;
#pragma unroll
                for (int ks = 0; ks < 2; ++ks) {
                    v16bf ak;
                    v8bf lo = *(const v8bf*)&Ks[kt*16 + ln][ks*32 + h*16];
                    v8bf hi = *(const v8bf*)&Ks[kt*16 + ln][ks*32 + h*16 + 8];
#pragma unroll
                    for (int i = 0; i < 8; ++i) { ak[i] = lo[i]; ak[i+8] = hi[i]; }
                    ct[kt] = __builtin_amdgcn_wmma_f32_16x16x32_bf16(false, ak, false, bq[ks], (short)0, ct[kt], false, false);
                }
            }

            // ---- additive key-padding mask (uniform per key row, broadcast LDS loads) ----
            float s[4][8];
#pragma unroll
            for (int kt = 0; kt < 4; ++kt) {
                v4f kaL = *(const v4f*)&kadd[kt*16 + h*8];
                v4f kaH = *(const v4f*)&kadd[kt*16 + h*8 + 4];
#pragma unroll
                for (int r = 0; r < 4; ++r) {
                    s[kt][r]   = ct[kt][r]   + kaL[r];
                    s[kt][r+4] = ct[kt][r+4] + kaH[r];
                }
            }

            // ---- causal mask: only the diagonal iteration needs per-element compares ----
            if (kb + KT - 1 > qmin_w) {
#pragma unroll
                for (int kt = 0; kt < 4; ++kt)
#pragma unroll
                    for (int r = 0; r < 8; ++r) {
                        const int key = kb + kt*16 + h*8 + r;
                        s[kt][r] = (key <= vq) ? s[kt][r] : MASKV;
                    }
            }

            // ---- in-lane max over 32 keys + one cross-half combine ----
            float mx = s[0][0];
#pragma unroll
            for (int kt = 0; kt < 4; ++kt)
#pragma unroll
                for (int r = 0; r < 8; ++r) mx = fmaxf(mx, s[kt][r]);
            mx = fmaxf(mx, __shfl_xor(mx, 16, 32));

            const float mn    = fmaxf(m_run, mx);
            const float alpha = __expf(m_run - mn);
            m_run = mn;

            float p[4][8];
            float rs = 0.f;
#pragma unroll
            for (int kt = 0; kt < 4; ++kt)
#pragma unroll
                for (int r = 0; r < 8; ++r) {
                    p[kt][r] = __expf(s[kt][r] - mn);
                    rs += p[kt][r];
                }
            rs += __shfl_xor(rs, 16, 32);
            l_run = l_run * alpha + rs;

#pragma unroll
            for (int t = 0; t < 4; ++t)
#pragma unroll
                for (int r = 0; r < 8; ++r) acc[t][r] *= alpha;

            // ---- build B-fragments of P^T: cross-half exchange, tiles (0,1) and (2,3) ----
            v16bf bp0, bp1;
#pragma unroll
            for (int r = 0; r < 8; ++r) {
                const float send0 = h ? p[0][r] : p[1][r];
                const float recv0 = __shfl_xor(send0, 16, 32);
                bp0[r]     = (__bf16)(h ? recv0   : p[0][r]);
                bp0[r + 8] = (__bf16)(h ? p[1][r] : recv0);
                const float send1 = h ? p[2][r] : p[3][r];
                const float recv1 = __shfl_xor(send1, 16, 32);
                bp1[r]     = (__bf16)(h ? recv1   : p[2][r]);
                bp1[r + 8] = (__bf16)(h ? p[3][r] : recv1);
            }

            // ---- O^T += V^T P^T : A = V^T rows (contiguous from transposed V) ----
#pragma unroll
            for (int t = 0; t < 4; ++t) {
                {
                    v16bf av;
                    v8bf lo = *(const v8bf*)&Vt[t*16 + ln][h*16];
                    v8bf hi = *(const v8bf*)&Vt[t*16 + ln][h*16 + 8];
#pragma unroll
                    for (int i = 0; i < 8; ++i) { av[i] = lo[i]; av[i+8] = hi[i]; }
                    acc[t] = __builtin_amdgcn_wmma_f32_16x16x32_bf16(false, av, false, bp0, (short)0, acc[t], false, false);
                }
                {
                    v16bf av;
                    v8bf lo = *(const v8bf*)&Vt[t*16 + ln][32 + h*16];
                    v8bf hi = *(const v8bf*)&Vt[t*16 + ln][40 + h*16];
#pragma unroll
                    for (int i = 0; i < 8; ++i) { av[i] = lo[i]; av[i+8] = hi[i]; }
                    acc[t] = __builtin_amdgcn_wmma_f32_16x16x32_bf16(false, av, false, bp1, (short)0, acc[t], false, false);
                }
            }
        }
    }

    // ---- epilogue: per-lane 1/l * qmask, transpose via LDS, coalesced stores ----
    {
        const float sc = ((l_run > 0.f) ? 1.f / l_run : 0.f) * qmask[wid*16 + ln];
#pragma unroll
        for (int t = 0; t < 4; ++t)
#pragma unroll
            for (int r = 0; r < 8; ++r)
                Os[wid][ln][t*16 + h*8 + r] = acc[t][r] * sc;

        asm volatile("" ::: "memory");  // same-wave LDS store->load ordering

        const float* osf = &Os[wid][0][0];
        float* og = Og + ((size_t)b * SEQ + q0 + wid*16) * DIM;
#pragma unroll
        for (int i = 0; i < 8; ++i) {
            const int c = (lane + i*32) * 4;
            *(v4f*)(og + c) = *(const v4f*)(osf + c);
        }
    }
}

extern "C" void kernel_launch(void* const* d_in, const int* in_sizes, int n_in,
                              void* d_out, int out_size, void* d_ws, size_t ws_size,
                              hipStream_t stream) {
    (void)in_sizes; (void)n_in; (void)out_size; (void)d_ws; (void)ws_size;
    const float* Q = (const float*)d_in[0];
    const float* K = (const float*)d_in[1];
    const float* V = (const float*)d_in[2];
    float* O = (float*)d_out;
    dim3 grid(BATCH * (SEQ / QT));
    dim3 block(128);
    hipLaunchKernelGGL(fa_fwd_kernel, grid, block, 0, stream, Q, K, V, O);
}
